// MyWeighter_10350871183799
// MI455X (gfx1250) — matrix operand order repaired and checked by reading, not compiled, not run on hardware
//
#include <hip/hip_runtime.h>
#include <hip/hip_bf16.h>

typedef __attribute__((ext_vector_type(16))) _Float16 v16h;
typedef __attribute__((ext_vector_type(8)))  float    v8f;

#define BATCH 8192
#define NCLS  4096
#define BINS  64
#define CPB   16           // classes per block (WMMA M dimension)

// ---------------------------------------------------------------------------
// Kernel 1: per-class histogram -> logit -> [16x64]@[64x64] MLP via WMMA ->
// softmax/cumsum -> 66 knots per class written to workspace.
// One block = 256 threads = 8 waves, handles 16 classes.
// ---------------------------------------------------------------------------
__global__ void __launch_bounds__(256)
fit_knots_kernel(const float* __restrict__ score,
                 const int*   __restrict__ part,
                 const float* __restrict__ W1,
                 const float* __restrict__ b1,
                 const float* __restrict__ W2,
                 const float* __restrict__ b2,
                 float* __restrict__ knots)
{
    __shared__ int      hist[CPB][BINS];
    __shared__ float    cntf[CPB];
    __shared__ _Float16 Hs [CPB][BINS];   // layer-1 input (logits), f16
    __shared__ _Float16 H2s[CPB][BINS];   // layer-2 input (leaky-relu out)
    __shared__ _Float16 W1s[BINS][BINS];  // W1[n][k] as given (row-major)
    __shared__ _Float16 W2s[BINS][BINS];
    __shared__ float    Z2s[CPB][BINS];   // layer-2 f32 output

    const int tid  = threadIdx.x;
    const int lane = tid & 31;
    const int wv   = tid >> 5;            // wave id 0..7
    const int c0   = blockIdx.x * CPB;    // first class of this block

    // --- stage weights (f32 -> f16) and zero histograms ---
    for (int i = tid; i < BINS * BINS; i += 256) {
        W1s[i >> 6][i & 63] = (_Float16)W1[i];
        W2s[i >> 6][i & 63] = (_Float16)W2[i];
    }
    for (int i = tid; i < CPB * BINS; i += 256)
        hist[i >> 6][i & 63] = 0;
    __syncthreads();

    // --- masked histogram: lanes 0..15 of each half-wave cover 16 classes ---
    {
        const int cl = tid & 15;                   // local class
        for (int r = (tid >> 4); r < BATCH; r += 16) {
            const int gi = r * NCLS + c0 + cl;
            if (part[gi] == 1) {
                float s = score[gi];
                int b = (int)(s * 64.0f);
                b = min(63, max(0, b));
                atomicAdd(&hist[cl][b], 1);
            }
        }
    }
    __syncthreads();

    // --- per-class count = total histogram mass (guard empty with 1) ---
    if (tid < CPB) {
        int s = 0;
        #pragma unroll
        for (int b = 0; b < BINS; ++b) s += hist[tid][b];
        cntf[tid] = fmaxf((float)s, 1.0f);
    }
    __syncthreads();

    // --- normalize, clip, logit -> f16 H ---
    for (int i = tid; i < CPB * BINS; i += 256) {
        const int m = i >> 6, bn = i & 63;
        float p = (float)hist[m][bn] / cntf[m];
        p = fminf(1.0f - 1e-6f, fmaxf(1e-6f, p));
        Hs[m][bn] = (_Float16)(logf(p) - log1pf(-p));
    }
    __syncthreads();

    const int half  = lane >> 4;          // 0/1: lane half
    const int nlane = lane & 15;          // N (and M for A-operand)
    const int n0    = (wv & 3) * 16;      // output column tile

    // --- layer 1: Z1 = H @ W1^T + b1, leaky-relu -> H2 (WMMA f16->f32) ---
    if (wv < 4) {
        v8f acc;
        const float bias = b1[n0 + nlane];
        #pragma unroll
        for (int j = 0; j < 8; ++j) acc[j] = bias;
        #pragma unroll
        for (int ks = 0; ks < 2; ++ks) {
            const int kbase = ks * 32;
            v16h A, Bv;
            #pragma unroll
            for (int v = 0; v < 8; ++v) {
                const int ka = (v < 4) ? (half * 8 + 2 * v)
                                       : (16 + half * 8 + 2 * (v - 4));
                A[2 * v]     = Hs[nlane][kbase + ka];
                A[2 * v + 1] = Hs[nlane][kbase + ka + 1];
                const int kb = (v < 4) ? (half * 8 + 2 * v)
                                       : (16 + half * 8 + 2 * (v - 4));
                Bv[2 * v]     = W1s[n0 + nlane][kbase + kb];
                Bv[2 * v + 1] = W1s[n0 + nlane][kbase + kb + 1];
            }
            acc = __builtin_amdgcn_wmma_f32_16x16x32_f16(
                      false, A, false, Bv, (short)0, acc, false, false);
        }
        #pragma unroll
        for (int v = 0; v < 8; ++v) {
            const int m = v + 8 * half;
            float z = acc[v];
            z = (z > 0.0f) ? z : 0.01f * z;
            H2s[m][n0 + nlane] = (_Float16)z;
        }
    }
    __syncthreads();

    // --- layer 2: Z2 = H2 @ W2^T + b2 (WMMA) ---
    if (wv < 4) {
        v8f acc;
        const float bias = b2[n0 + nlane];
        #pragma unroll
        for (int j = 0; j < 8; ++j) acc[j] = bias;
        #pragma unroll
        for (int ks = 0; ks < 2; ++ks) {
            const int kbase = ks * 32;
            v16h A, Bv;
            #pragma unroll
            for (int v = 0; v < 8; ++v) {
                const int ka = (v < 4) ? (half * 8 + 2 * v)
                                       : (16 + half * 8 + 2 * (v - 4));
                A[2 * v]     = H2s[nlane][kbase + ka];
                A[2 * v + 1] = H2s[nlane][kbase + ka + 1];
                const int kb = (v < 4) ? (half * 8 + 2 * v)
                                       : (16 + half * 8 + 2 * (v - 4));
                Bv[2 * v]     = W2s[n0 + nlane][kbase + kb];
                Bv[2 * v + 1] = W2s[n0 + nlane][kbase + kb + 1];
            }
            acc = __builtin_amdgcn_wmma_f32_16x16x32_f16(
                      false, A, false, Bv, (short)0, acc, false, false);
        }
        #pragma unroll
        for (int v = 0; v < 8; ++v)
            Z2s[v + 8 * half][n0 + nlane] = acc[v];
    }
    __syncthreads();

    // --- softmax + cumsum -> 66 knots per class ---
    if (tid < CPB) {
        const int c = c0 + tid;
        float mx = -1e30f;
        #pragma unroll
        for (int i = 0; i < BINS; ++i) mx = fmaxf(mx, Z2s[tid][i]);
        float ssum = 0.0f;
        #pragma unroll
        for (int i = 0; i < BINS; ++i) ssum += __expf(Z2s[tid][i] - mx);
        const float inv = 1.0f / ssum;
        float* kr = knots + (size_t)c * 66;
        kr[0] = 0.0f;
        float cum = 0.0f;
        #pragma unroll
        for (int i = 0; i < BINS; ++i) {
            cum += __expf(Z2s[tid][i] - mx) * inv;
            kr[1 + i] = cum;
        }
        kr[65] = 1.0f;
    }
}

// ---------------------------------------------------------------------------
// Kernel 2: streaming piecewise-linear interpolation, float4 per thread.
// ---------------------------------------------------------------------------
__global__ void __launch_bounds__(256)
interp_kernel(const float4* __restrict__ s4,
              const int4*   __restrict__ m4,
              const float*  __restrict__ knots,
              float4* __restrict__ out4,
              int n4)
{
    const int i = blockIdx.x * blockDim.x + threadIdx.x;
    if (i >= n4) return;

    const float4 sv = s4[i];
    const int4   mv = m4[i];
    const int cbase = (i * 4) & (NCLS - 1);   // NCLS divisible by 4 -> same row

    const float s[4] = {sv.x, sv.y, sv.z, sv.w};
    const int   m[4] = {mv.x, mv.y, mv.z, mv.w};
    float r[4];

    #pragma unroll
    for (int j = 0; j < 4; ++j) {
        const float sc = s[j];
        // searchsorted(x_knots, sc, 'left') - 1, clipped to [0, 64]
        int idx = (int)ceilf(sc * 64.0f + 0.5f) - 1;
        idx = min(64, max(0, idx));
        // x_knots[k] = clamp((k - 0.5)/64, 0, 1)
        const float x0 = fminf(1.0f, fmaxf(0.0f, ((float)idx - 0.5f) * (1.0f / 64.0f)));
        const float x1 = fminf(1.0f, fmaxf(0.0f, ((float)idx + 0.5f) * (1.0f / 64.0f)));
        const float* kr = knots + (size_t)(cbase + j) * 66;
        const float y0 = kr[idx];
        const float y1 = kr[idx + 1];
        const float w  = y0 + (y1 - y0) / (x1 - x0) * (sc - x0);
        r[j] = (m[j] == 1) ? w : 1.0f;
    }
    out4[i] = make_float4(r[0], r[1], r[2], r[3]);
}

extern "C" void kernel_launch(void* const* d_in, const int* in_sizes, int n_in,
                              void* d_out, int out_size, void* d_ws, size_t ws_size,
                              hipStream_t stream) {
    const float* score = (const float*)d_in[0];
    const int*   part  = (const int*)  d_in[1];
    const float* W1    = (const float*)d_in[2];
    const float* b1    = (const float*)d_in[3];
    const float* W2    = (const float*)d_in[4];
    const float* b2    = (const float*)d_in[5];
    float* out   = (float*)d_out;
    float* knots = (float*)d_ws;                 // NCLS*66 floats ~ 1.05 MB

    // Pass 1: histogram + MLP (WMMA) + knots. 4096/16 = 256 blocks.
    fit_knots_kernel<<<NCLS / CPB, 256, 0, stream>>>(
        score, part, W1, b1, W2, b2, knots);

    // Pass 2: interpolation. 8192*4096/4 float4 elements.
    const int n4 = (BATCH * NCLS) / 4;
    interp_kernel<<<(n4 + 255) / 256, 256, 0, stream>>>(
        (const float4*)score, (const int4*)part, knots, (float4*)out, n4);
}
